// FusedMoE_87686052315721
// MI455X (gfx1250) — matrix operand order repaired and checked by reading, not compiled, not run on hardware
//
#include <hip/hip_runtime.h>
#include <hip/hip_bf16.h>
#include <math.h>

#define TOK   8192
#define NEXP  8
#define HID   2048
#define INTER 5632

typedef __attribute__((ext_vector_type(16))) __bf16       v16bf;
typedef __attribute__((ext_vector_type(8)))  float        v8f;
typedef __attribute__((ext_vector_type(4)))  unsigned int u32x4;
typedef __attribute__((ext_vector_type(4)))  float        f32x4;

union FragBF { v16bf v; u32x4 q[2]; };

static __device__ __forceinline__ unsigned short f2bf(float f) {
    union { float f; unsigned int u; } x; x.f = f;
    unsigned int r = x.u + 0x7FFFu + ((x.u >> 16) & 1u);   // RNE
    return (unsigned short)(r >> 16);
}

#if __has_builtin(__builtin_amdgcn_cvt_pk_bf16_f32)
typedef __attribute__((ext_vector_type(2))) __bf16 v2bf;
static __device__ __forceinline__ unsigned int pk2(float a, float b) {
    union { v2bf v; unsigned int u; } c;
    c.v = __builtin_amdgcn_cvt_pk_bf16_f32(a, b);          // v_cvt_pk_bf16_f32
    return c.u;
}
#else
static __device__ __forceinline__ unsigned int pk2(float a, float b) {
    return (unsigned int)f2bf(a) | ((unsigned int)f2bf(b) << 16);
}
#endif

static __device__ __forceinline__ u32x4 pk8(f32x4 a, f32x4 b) {
    u32x4 p; p[0] = pk2(a[0], a[1]); p[1] = pk2(a[2], a[3]);
    p[2] = pk2(b[0], b[1]); p[3] = pk2(b[2], b[3]);
    return p;
}
static __device__ __forceinline__ v8f vzero() {
    v8f z; for (int i = 0; i < 8; ++i) z[i] = 0.f; return z;
}

// fast SiLU: output is bf16-rounded right after, so native exp/rcp suffice
static __device__ __forceinline__ float silu_fast(float g) {
    return __fdividef(g, 1.f + __expf(-g));   // v_exp_f32 + v_rcp_f32 path
}

// ---------------- kernel 1: top-2 routing + softmax -------------------------
__global__ void moe_route(const float* __restrict__ logits,
                          int* __restrict__ tk_id, float* __restrict__ tk_w) {
    int t = blockIdx.x * blockDim.x + threadIdx.x;
    if (t >= TOK) return;
    float l[NEXP];
#pragma unroll
    for (int j = 0; j < NEXP; ++j) l[j] = logits[t * NEXP + j];
    float b0 = -1e30f; int i0 = 0;
#pragma unroll
    for (int j = 0; j < NEXP; ++j) if (l[j] > b0) { b0 = l[j]; i0 = j; }
    float b1 = -1e30f; int i1 = 0;
#pragma unroll
    for (int j = 0; j < NEXP; ++j) if (j != i0 && l[j] > b1) { b1 = l[j]; i1 = j; }
    float e1 = expf(b1 - b0);           // b0 is max
    float inv = 1.f / (1.f + e1);
    tk_id[2 * t + 0] = i0; tk_w[2 * t + 0] = inv;
    tk_id[2 * t + 1] = i1; tk_w[2 * t + 1] = e1 * inv;
}

// ---------------- kernel 2: deterministic per-expert compaction -------------
__global__ void moe_compact(const int* __restrict__ tk_id,
                            const float* __restrict__ tk_w,
                            int* __restrict__ tok_list, float* __restrict__ tok_wgt,
                            int* __restrict__ hdr) {
    int tid = threadIdx.x;
    if (tid < NEXP) {
        int e = tid, c = 0;
        for (int t = 0; t < TOK; ++t)
            for (int k = 0; k < 2; ++k)
                if (tk_id[2 * t + k] == e) {
                    tok_list[e * TOK + c] = t;
                    tok_wgt[e * TOK + c] = tk_w[2 * t + k];
                    ++c;
                }
        hdr[e] = c;
    }
    __syncthreads();
    if (tid == 0) {                      // 128-row M-tile prefix offsets
        int off = 0; hdr[8] = 0;
        for (int e = 0; e < NEXP; ++e) { off += (hdr[e] + 127) >> 7; hdr[9 + e] = off; }
        hdr[17] = off;                   // total M tiles
    }
}

// ---------------- kernel 3: fp32 -> bf16 streaming convert ------------------
// used for x (once) and the three weight tensors (read-once -> NT loads)
__global__ void moe_cvt(const float* __restrict__ src, unsigned short* __restrict__ dst) {
    size_t i = ((size_t)blockIdx.x * blockDim.x + threadIdx.x) * 8;
    f32x4 a = __builtin_nontemporal_load((const f32x4*)(src + i));
    f32x4 b = __builtin_nontemporal_load((const f32x4*)(src + i + 4));
    *(u32x4*)(dst + i) = pk8(a, b);
}

// fragment loaders (ISA 7.12.2 16-bit layouts), LDS tiles are [rows][32] bf16
static __device__ __forceinline__ FragBF loadA(const unsigned short* lds, int row0, int lane) {
    FragBF f;
    int row = row0 + (lane & 15), half = lane >> 4;
    const u32x4* p = (const u32x4*)(lds + row * 32 + half * 8);  // K 0-7 | 8-15
    f.q[0] = p[0];                                               // +32B: K 16-23 | 24-31
    f.q[1] = p[2];
    return f;
}
static __device__ __forceinline__ FragBF loadB(const unsigned short* lds, int row0, int lane) {
    FragBF f;
    int row = row0 + (lane & 15), half = lane >> 4;
    const u32x4* p = (const u32x4*)(lds + row * 32 + half * 8);  // K 0-7 | 8-15, +32B: K 16-23 | 24-31
    f.q[0] = p[0];
    f.q[1] = p[2];
    return f;
}

// ---------------- kernel 4: grouped GEMM1 (gate/up) + SiLU*mul --------------
// per job: expert e, M-tile (128 gathered tokens), N-tile (64 cols of I)
// all-bf16 operands; 2-stage LDS pipeline, 1 A-chunk + 1 B-chunk staged/thread
__global__ void __launch_bounds__(256, 1)
moe_gemm1(const unsigned short* __restrict__ xbf,
          const unsigned short* __restrict__ w1bf, const unsigned short* __restrict__ w3bf,
          unsigned short* __restrict__ act,
          const int* __restrict__ tok_list,
          const int* __restrict__ hdr) {
    __shared__ unsigned short ldsA[2][128 * 32];
    __shared__ unsigned short ldsB[2][2][64 * 32];   // [stage][gate|up]

    const int NT = INTER / 64;                 // 88
    const int totalJobs = hdr[17] * NT;
    const int lane = threadIdx.x & 31, wave = threadIdx.x >> 5;
    const int waveM = wave & 3, waveN = wave >> 2;
    const int rA = threadIdx.x >> 1, hA = threadIdx.x & 1;   // A: 256 x 32B chunks
    const int gu = threadIdx.x >> 7;                         // 0: gate tile, 1: up tile
    const int rB = (threadIdx.x & 127) >> 1, hB = threadIdx.x & 1;

    for (int job = blockIdx.x; job < totalJobs; job += gridDim.x) {
        const int mtG = job / NT, nT = job % NT, n0 = nT * 64;
        int e = 0;
        while (mtG >= hdr[9 + e]) ++e;
        const int mLocal = mtG - hdr[8 + e];
        const int cnt = hdr[e];
        const int padBase = hdr[8 + e] * 128;

        // hoisted gather + source pointers (job-invariant)
        const int pidx = mLocal * 128 + rA;
        const int tok = (pidx < cnt) ? tok_list[e * TOK + pidx] : 0;
        const unsigned short* aSrc = xbf + (size_t)tok * HID + hA * 16;
        const unsigned short* bSrc = (gu ? w3bf : w1bf) +
            (size_t)e * INTER * HID + (size_t)(n0 + rB) * HID + hB * 16;

        v8f accG[2][2], accU[2][2];
#pragma unroll
        for (int i = 0; i < 2; ++i)
#pragma unroll
            for (int j = 0; j < 2; ++j) { accG[i][j] = vzero(); accU[i][j] = vzero(); }

        // prologue: stage 0
        u32x4 ra0 = *(const u32x4*)aSrc;
        u32x4 ra1 = *(const u32x4*)(aSrc + 8);
        u32x4 rb0 = *(const u32x4*)bSrc;
        u32x4 rb1 = *(const u32x4*)(bSrc + 8);
        { u32x4* d = (u32x4*)&ldsA[0][rA * 32 + hA * 16];     d[0] = ra0; d[1] = ra1; }
        { u32x4* d = (u32x4*)&ldsB[0][gu][rB * 32 + hB * 16]; d[0] = rb0; d[1] = rb1; }

        const int NKI = HID / 32;              // 64
        for (int i = 0; i < NKI; ++i) {
            const int cur = i & 1, nxt = cur ^ 1;
            const int ktn = (i + 1) * 32;
            if (i + 1 < NKI) {                 // issue next-stage loads (in flight over WMMA)
                ra0 = *(const u32x4*)(aSrc + ktn);
                ra1 = *(const u32x4*)(aSrc + ktn + 8);
                rb0 = *(const u32x4*)(bSrc + ktn);
                rb1 = *(const u32x4*)(bSrc + ktn + 8);
                __builtin_prefetch(bSrc + ktn + 32, 0, 1);
            }
            __syncthreads();                   // stage `cur` visible

            FragBF fa[2], fg[2], fu[2];
#pragma unroll
            for (int sm = 0; sm < 2; ++sm) fa[sm] = loadA(ldsA[cur], waveM * 32 + sm * 16, lane);
#pragma unroll
            for (int sn = 0; sn < 2; ++sn) {
                fg[sn] = loadB(ldsB[cur][0], waveN * 32 + sn * 16, lane);
                fu[sn] = loadB(ldsB[cur][1], waveN * 32 + sn * 16, lane);
            }
#pragma unroll
            for (int sm = 0; sm < 2; ++sm)
#pragma unroll
                for (int sn = 0; sn < 2; ++sn) {
                    accG[sm][sn] = __builtin_amdgcn_wmma_f32_16x16x32_bf16(
                        false, fa[sm].v, false, fg[sn].v, (short)0, accG[sm][sn], false, false);
                    accU[sm][sn] = __builtin_amdgcn_wmma_f32_16x16x32_bf16(
                        false, fa[sm].v, false, fu[sn].v, (short)0, accU[sm][sn], false, false);
                }

            if (i + 1 < NKI) {                 // park next stage after compute
                u32x4* d = (u32x4*)&ldsA[nxt][rA * 32 + hA * 16];      d[0] = ra0; d[1] = ra1;
                u32x4* d2 = (u32x4*)&ldsB[nxt][gu][rB * 32 + hB * 16]; d2[0] = rb0; d2[1] = rb1;
            }
        }

        // epilogue: SiLU(gate)*up -> bf16 act (streamed, non-temporal)
#pragma unroll
        for (int sm = 0; sm < 2; ++sm)
#pragma unroll
            for (int sn = 0; sn < 2; ++sn)
#pragma unroll
                for (int v = 0; v < 8; ++v) {
                    float h = silu_fast(accG[sm][sn][v]) * accU[sm][sn][v];
                    int row = waveM * 32 + sm * 16 + (lane >> 4) * 8 + v;
                    int col = n0 + waveN * 32 + sn * 16 + (lane & 15);
                    size_t arow = (size_t)(padBase + mLocal * 128 + row);
                    __builtin_nontemporal_store(f2bf(h), act + arow * INTER + col);
                }
    }
}

// ---------------- kernel 5: grouped GEMM2 (down proj) + weighted scatter ----
__global__ void __launch_bounds__(256, 1)
moe_gemm2(const unsigned short* __restrict__ act,
          const unsigned short* __restrict__ w2bf,
          float* __restrict__ out,
          const int* __restrict__ tok_list,
          const float* __restrict__ tok_wgt,
          const int* __restrict__ hdr) {
    __shared__ unsigned short ldsA[2][128 * 32];
    __shared__ unsigned short ldsB[2][128 * 32];

    const int NT = HID / 128;                  // 16
    const int totalJobs = hdr[17] * NT;
    const int lane = threadIdx.x & 31, wave = threadIdx.x >> 5;
    const int waveM = wave & 3, waveN = wave >> 2;
    const int rA = threadIdx.x >> 1, hA = threadIdx.x & 1;

    for (int job = blockIdx.x; job < totalJobs; job += gridDim.x) {
        const int mtG = job / NT, nT = job % NT, n0 = nT * 128;
        int e = 0;
        while (mtG >= hdr[9 + e]) ++e;
        const int mLocal = mtG - hdr[8 + e];
        const int cnt = hdr[e];
        const int padBase = hdr[8 + e] * 128;

        const unsigned short* aSrc =
            act + (size_t)(padBase + mLocal * 128 + rA) * INTER + hA * 16;
        const unsigned short* bSrc =
            w2bf + (size_t)e * HID * INTER + (size_t)(n0 + rA) * INTER + hA * 16;

        v8f acc[2][4];
#pragma unroll
        for (int i = 0; i < 2; ++i)
#pragma unroll
            for (int j = 0; j < 4; ++j) acc[i][j] = vzero();

        // prologue: stage 0 (act is stream-once -> non-temporal)
        u32x4 ra0 = __builtin_nontemporal_load((const u32x4*)aSrc);
        u32x4 ra1 = __builtin_nontemporal_load((const u32x4*)(aSrc + 8));
        u32x4 rb0 = *(const u32x4*)bSrc;
        u32x4 rb1 = *(const u32x4*)(bSrc + 8);
        { u32x4* d = (u32x4*)&ldsA[0][rA * 32 + hA * 16]; d[0] = ra0; d[1] = ra1; }
        { u32x4* d = (u32x4*)&ldsB[0][rA * 32 + hA * 16]; d[0] = rb0; d[1] = rb1; }

        const int NKI = INTER / 32;            // 176
        for (int i = 0; i < NKI; ++i) {
            const int cur = i & 1, nxt = cur ^ 1;
            const int ktn = (i + 1) * 32;
            if (i + 1 < NKI) {
                ra0 = __builtin_nontemporal_load((const u32x4*)(aSrc + ktn));
                ra1 = __builtin_nontemporal_load((const u32x4*)(aSrc + ktn + 8));
                rb0 = *(const u32x4*)(bSrc + ktn);
                rb1 = *(const u32x4*)(bSrc + ktn + 8);
                __builtin_prefetch(bSrc + ktn + 32, 0, 1);
            }
            __syncthreads();

            FragBF fa[2], fb[4];
#pragma unroll
            for (int sm = 0; sm < 2; ++sm) fa[sm] = loadA(ldsA[cur], waveM * 32 + sm * 16, lane);
#pragma unroll
            for (int sn = 0; sn < 4; ++sn) fb[sn] = loadB(ldsB[cur], waveN * 64 + sn * 16, lane);
#pragma unroll
            for (int sm = 0; sm < 2; ++sm)
#pragma unroll
                for (int sn = 0; sn < 4; ++sn)
                    acc[sm][sn] = __builtin_amdgcn_wmma_f32_16x16x32_bf16(
                        false, fa[sm].v, false, fb[sn].v, (short)0, acc[sm][sn], false, false);

            if (i + 1 < NKI) {
                u32x4* d = (u32x4*)&ldsA[nxt][rA * 32 + hA * 16]; d[0] = ra0; d[1] = ra1;
                u32x4* d2 = (u32x4*)&ldsB[nxt][rA * 32 + hA * 16]; d2[0] = rb0; d2[1] = rb1;
            }
        }

        // epilogue: scale by router weight, scatter-add into out[token, n]
#pragma unroll
        for (int sm = 0; sm < 2; ++sm)
#pragma unroll
            for (int v = 0; v < 8; ++v) {
                int row = waveM * 32 + sm * 16 + (lane >> 4) * 8 + v;
                int pidx = mLocal * 128 + row;
                bool valid = pidx < cnt;
                int tok = valid ? tok_list[e * TOK + pidx] : 0;
                float w = valid ? tok_wgt[e * TOK + pidx] : 0.f;
#pragma unroll
                for (int sn = 0; sn < 4; ++sn) {
                    int col = n0 + waveN * 64 + sn * 16 + (lane & 15);
                    atomicAdd(out + (size_t)tok * HID + col, acc[sm][sn][v] * w);
                }
            }
    }
}

extern "C" void kernel_launch(void* const* d_in, const int* in_sizes, int n_in,
                              void* d_out, int out_size, void* d_ws, size_t ws_size,
                              hipStream_t stream) {
    const float* x      = (const float*)d_in[0];
    const float* logits = (const float*)d_in[1];
    const float* w1     = (const float*)d_in[2];
    const float* w2     = (const float*)d_in[3];
    const float* w3     = (const float*)d_in[4];
    float* out = (float*)d_out;

    const size_t WELEM = (size_t)NEXP * INTER * HID;   // elements per weight tensor

    // workspace carve-up
    char* ws = (char*)d_ws;
    size_t off = 0;
    int*   hdr      = (int*)(ws + off);                off += 256;
    int*   tk_id    = (int*)(ws + off);                off += (size_t)2 * TOK * 4;
    float* tk_w     = (float*)(ws + off);              off += (size_t)2 * TOK * 4;
    int*   tok_list = (int*)(ws + off);                off += (size_t)NEXP * TOK * 4;
    float* tok_wgt  = (float*)(ws + off);              off += (size_t)NEXP * TOK * 4;
    unsigned short* xbf  = (unsigned short*)(ws + off); off += (size_t)TOK * HID * 2;
    unsigned short* w1bf = (unsigned short*)(ws + off); off += WELEM * 2;
    unsigned short* w3bf = (unsigned short*)(ws + off); off += WELEM * 2;
    unsigned short* w2bf = (unsigned short*)(ws + off); off += WELEM * 2;
    unsigned short* act  = (unsigned short*)(ws + off); // padded [rows, INTER] bf16

    hipMemsetAsync(d_out, 0, (size_t)TOK * HID * sizeof(float), stream);

    moe_route  <<<(TOK + 255) / 256, 256, 0, stream>>>(logits, tk_id, tk_w);
    moe_compact<<<1, 64, 0, stream>>>(tk_id, tk_w, tok_list, tok_wgt, hdr);
    moe_cvt    <<<(TOK * HID / 8) / 256, 256, 0, stream>>>(x, xbf);
    moe_cvt    <<<(int)(WELEM / 8 / 256), 256, 0, stream>>>(w1, w1bf);
    moe_cvt    <<<(int)(WELEM / 8 / 256), 256, 0, stream>>>(w3, w3bf);
    moe_cvt    <<<(int)(WELEM / 8 / 256), 256, 0, stream>>>(w2, w2bf);
    moe_gemm1  <<<2048, 256, 0, stream>>>(xbf, w1bf, w3bf, act, tok_list, hdr);
    moe_gemm2  <<<1024, 256, 0, stream>>>(act, w2bf, out, tok_list, tok_wgt, hdr);
}
